// RNN_13623636263694
// MI455X (gfx1250) — matrix operand order repaired and checked by reading, not compiled, not run on hardware
//
#include <hip/hip_runtime.h>
#include <hip/hip_bf16.h>

typedef __bf16 bf16;
typedef __attribute__((ext_vector_type(16))) __bf16 v16bf;
typedef __attribute__((ext_vector_type(8)))  __bf16 v8bf;
typedef __attribute__((ext_vector_type(8)))  float   v8f;
typedef __attribute__((ext_vector_type(4)))  float   v4f;
typedef __attribute__((ext_vector_type(4)))  unsigned v4u;
typedef __attribute__((ext_vector_type(8)))  int      v8i;
typedef __attribute__((ext_vector_type(4)))  int      v4i;

namespace {
constexpr int B_ = 64, S_ = 2048, I_ = 512, H_ = 512;
constexpr int LDSW = 528;        // h-buffer pitch: 512 + 16 bf16 pad (1056B)
constexpr int BPITCH = 40;       // xproj B-tile pitch: 32 + 8 bf16 pad (80B)

// workspace byte offsets
constexpr size_t OFF_WIH  = 0;                                  // H*I bf16
constexpr size_t OFF_WHH  = OFF_WIH + (size_t)H_ * I_ * 2;      // H*H bf16
constexpr size_t OFF_HB0  = OFF_WHH + (size_t)H_ * H_ * 2;      // B*H bf16
constexpr size_t OFF_BSUM = OFF_HB0 + (size_t)B_ * H_ * 2;      // H f32
}

// 2-D bf16 tensor tile -> LDS via the Tensor Data Mover.
// Bitfields per CDNA5 ISA D# (group0: count/lds/global/type; group1: data_size,
// pad, tensor dims, tile dims, dim0 stride). Groups 2/3 unused for 2-D tiles.
__device__ __forceinline__ void tdm_load_2d(unsigned lds_off, const bf16* gptr,
                                            unsigned ten_d0, unsigned ten_d1,
                                            unsigned til_d0, unsigned til_d1,
                                            unsigned stride0,
                                            unsigned padi, unsigned pada)
{
    const unsigned long long ga = (unsigned long long)(uintptr_t)gptr;
    v4u g0;
    g0[0] = 1u;                                              // count=1, user desc
    g0[1] = lds_off;                                         // lds_addr (bytes)
    g0[2] = (unsigned)ga;                                    // global_addr[31:0]
    g0[3] = (unsigned)((ga >> 32) & 0x1FFFFFFu) | (2u << 30); // [56:32] | type=2
    v8i g1;
    unsigned w0 = (1u << 16);                                // data_size = 2 bytes
    if (pada) w0 |= (1u << 20) | (padi << 22) | (pada << 25);
    g1[0] = (int)w0;
    g1[1] = (int)((ten_d0 & 0xFFFFu) << 16);                 // tensor_dim0 lo16
    g1[2] = (int)((ten_d0 >> 16) | ((ten_d1 & 0xFFFFu) << 16));
    g1[3] = (int)((ten_d1 >> 16) | (til_d0 << 16));          // tile_dim0
    g1[4] = (int)(til_d1 & 0xFFFFu);                         // tile_dim1, tile_dim2=0
    g1[5] = (int)stride0;                                    // dim0 stride (elems)
    g1[6] = 0;
    g1[7] = 0;
    v4i z4 = {}; v8i z8 = {};
    __builtin_amdgcn_tensor_load_to_lds(g0, g1, z4, z4, z8, 0);
}

// ---------------------------------------------------------------- phase 0
__global__ void rnn_prep(const float* __restrict__ wih, const float* __restrict__ whh,
                         const float* __restrict__ h0,  const float* __restrict__ bih,
                         const float* __restrict__ bhh,
                         bf16* __restrict__ wih_bf, bf16* __restrict__ whh_bf,
                         bf16* __restrict__ hb0, float* __restrict__ bsum)
{
    int idx = blockIdx.x * blockDim.x + threadIdx.x;
    if (idx < H_ * I_) wih_bf[idx] = (bf16)wih[idx];
    if (idx < H_ * H_) whh_bf[idx] = (bf16)whh[idx];
    if (idx < B_ * H_) hb0[idx]    = (bf16)h0[idx];
    if (idx < H_)      bsum[idx]   = bih[idx] + bhh[idx];
}

// ---------------------------------------------------------------- phase 1
// out[m, n] = sum_k x[m,k] * Wih[n,k] + bsum[n]
// Block = 8 waves; wave computes 16(M) x 128(N). Grid = (N fast, M slow) so
// concurrent blocks share A rows -> A streams from HBM once (dominant 256MB).
// B tile (128 cols x 32 K, 8KB) is double-buffered in LDS via TDM; the DMA for
// k-step ks+1 overlaps the 8 WMMAs of ks, so no global-load waits in the loop.
__global__ void __launch_bounds__(256)
rnn_xproj(const float* __restrict__ x, const bf16* __restrict__ wih,
          const float* __restrict__ bsum, float* __restrict__ out)
{
    __shared__ bf16 sB[2][128 * BPITCH];   // 2 x 10KB, 80B pitch per B-column

    const int tid  = threadIdx.x;
    const int lane = tid & 31;        // wave32
    const int w    = tid >> 5;
    const int col  = lane & 15;
    const int rh   = lane >> 4;       // row-half select
    const int koffA = rh * 8;         // A: lanes>=16 hold K 8..15 / 24..31
    const int koffB = rh * 16;        // unused: B reads use koffA, lanes>=16 hold K 8..15 / 24..31

    const size_t m0 = (size_t)blockIdx.y * 128 + (size_t)w * 16;
    const int    n0 = blockIdx.x * 128;

    const float* arow = x + (m0 + (size_t)col) * I_;
    const bf16*  bbase = wih + (size_t)n0 * I_;

    if (w == 0) {   // prime buffer 0 with k-step 0
        tdm_load_2d((unsigned)(uintptr_t)&sB[0][0], bbase,
                    I_, 128, 32, 128, I_, /*padi 16dw*/ 3, /*pada 4dw*/ 3);
        __builtin_amdgcn_s_wait_tensorcnt(0);
    }
    __syncthreads();

    v8f acc[8] = {};

    for (int ks = 0; ks < 16; ++ks) {
        const int k0  = ks * 32;
        const int buf = ks & 1;

        if (w == 0 && ks < 15)   // DMA next k-slice into the other buffer
            tdm_load_2d((unsigned)(uintptr_t)&sB[buf ^ 1][0], bbase + (k0 + 32),
                        I_, 128, 32, 128, I_, 3, 3);

        // A fragment: f32 load, cvt->bf16, ISA lane layout
        v16bf A;
        {
            const v4f q0 = *(const v4f*)(arow + k0 + koffA);
            const v4f q1 = *(const v4f*)(arow + k0 + koffA + 4);
            const v4f q2 = *(const v4f*)(arow + k0 + koffA + 16);
            const v4f q3 = *(const v4f*)(arow + k0 + koffA + 20);
#pragma unroll
            for (int e = 0; e < 4; ++e) {
                A[e]      = (bf16)q0[e];
                A[4 + e]  = (bf16)q1[e];
                A[8 + e]  = (bf16)q2[e];
                A[12 + e] = (bf16)q3[e];
            }
        }

        // 8 WMMAs, B fragments straight from LDS
#pragma unroll
        for (int nt = 0; nt < 8; ++nt) {
            const int lr = nt * 16 + col;           // LDS row = local B column
            const v8bf b0 = *(const v8bf*)&sB[buf][lr * BPITCH + koffA];
            const v8bf b1 = *(const v8bf*)&sB[buf][lr * BPITCH + koffA + 16];
            v16bf Bm;
#pragma unroll
            for (int e = 0; e < 8; ++e) { Bm[e] = b0[e]; Bm[8 + e] = b1[e]; }
            acc[nt] = __builtin_amdgcn_wmma_f32_16x16x32_bf16(
                false, A, false, Bm, (short)0, acc[nt], false, false);
        }

        if (w == 0 && ks < 15) __builtin_amdgcn_s_wait_tensorcnt(0);
        __syncthreads();
    }

#pragma unroll
    for (int nt = 0; nt < 8; ++nt) {
        const int n = n0 + nt * 16 + col;
        const float bs = bsum[n];
#pragma unroll
        for (int r = 0; r < 8; ++r) {
            const size_t m = m0 + (size_t)(rh * 8 + r);
            out[m * H_ + n] = acc[nt][r] + bs;
        }
    }
}

// ---------------------------------------------------------------- phase 2
// h_t = relu(xproj_t + h_{t-1} @ Whh^T), in place over out.
// KEY: batch rows are independent -> the 64-row scan splits into 4 independent
// 16-row scans. One WG (1024 thr = 32 waves = 1 full WGP) per batch tile:
//   - h (16x512 bf16) ping-pongs in LDS; per-step sync is just s_barrier.
//   - wave w owns N-tile w: 16 chained WMMAs/step, Whh rows streamed from L2.
//   - NO device-wide synchronization, no atomics, no spin loops.
__global__ void __launch_bounds__(1024)
rnn_scan(float* __restrict__ out, const bf16* __restrict__ whh_bf,
         const bf16* __restrict__ hb0)
{
    __shared__ bf16 sH[2][16 * LDSW];   // 2 x 16.5KB h buffers, 1056B pitch

    const int g    = blockIdx.x;        // batch tile: rows 16g .. 16g+15
    const int tid  = threadIdx.x;
    const int lane = tid & 31;
    const int w    = tid >> 5;          // 32 waves -> 32 N-tiles
    const int col  = lane & 15;
    const int rh   = lane >> 4;
    const int koffA = rh * 8;
    const int koffB = rh * 16;
    const int mA   = col;               // local batch row feeding A
    const int ng   = w * 16 + col;      // global H column of this lane

    // B rows for this wave's N-tile: Whh row ng, K contiguous
    const bf16* brow = whh_bf + (size_t)ng * H_;

    if (w == 0) {   // TDM: h0 tile (16 x 512, stride 512) -> LDS buffer 0
        tdm_load_2d((unsigned)(uintptr_t)&sH[0][0], hb0 + (size_t)g * 16 * H_,
                    H_, 16, H_, 16, H_, /*padi 256dw*/ 7, /*pada 8dw*/ 7);
        __builtin_amdgcn_s_wait_tensorcnt(0);
    }
    __syncthreads();

    for (int t = 0; t < S_; ++t) {
        const bf16* cur = sH[t & 1];
        bf16*       nxt = sH[(t + 1) & 1];

        // pull NEXT step's x_proj line toward the caches while we compute
        const int tn = (t + 1 < S_) ? t + 1 : t;
        __builtin_prefetch(&out[(((size_t)(g * 16 + rh * 8)) * S_ + tn) * H_ + ng], 0, 1);

        // two accumulator chains (even/odd k-steps) to overlap WMMA latency
        v8f acc0 = {}, acc1 = {};
        for (int ks = 0; ks < 8; ++ks) {
            const int ka = ks * 64, kb = ks * 64 + 32;
            const v8bf a0 = *(const v8bf*)&cur[mA * LDSW + ka + koffA];
            const v8bf a1 = *(const v8bf*)&cur[mA * LDSW + ka + koffA + 16];
            const v8bf a2 = *(const v8bf*)&cur[mA * LDSW + kb + koffA];
            const v8bf a3 = *(const v8bf*)&cur[mA * LDSW + kb + koffA + 16];
            const v8bf b0 = *(const v8bf*)(brow + ka + koffA);
            const v8bf b1 = *(const v8bf*)(brow + ka + koffA + 16);
            const v8bf b2 = *(const v8bf*)(brow + kb + koffA);
            const v8bf b3 = *(const v8bf*)(brow + kb + koffA + 16);
            v16bf A0, B0, A1, B1;
#pragma unroll
            for (int e = 0; e < 8; ++e) {
                A0[e] = a0[e]; A0[8 + e] = a1[e];
                A1[e] = a2[e]; A1[8 + e] = a3[e];
                B0[e] = b0[e]; B0[8 + e] = b1[e];
                B1[e] = b2[e]; B1[8 + e] = b3[e];
            }
            acc0 = __builtin_amdgcn_wmma_f32_16x16x32_bf16(
                false, A0, false, B0, (short)0, acc0, false, false);
            acc1 = __builtin_amdgcn_wmma_f32_16x16x32_bf16(
                false, A1, false, B1, (short)0, acc1, false, false);
        }
        const v8f acc = acc0 + acc1;

        // epilogue: h = relu(xp + acc); f32 out (in place), bf16 h -> LDS nxt
#pragma unroll
        for (int r = 0; r < 8; ++r) {
            const int lr = rh * 8 + r;                 // local row in tile
            const int b  = g * 16 + lr;                // global batch row
            const size_t o = ((size_t)b * S_ + t) * H_ + ng;
            float v = out[o] + acc[r];
            v = v > 0.0f ? v : 0.0f;
            out[o] = v;
            nxt[lr * LDSW + ng] = (bf16)v;
            if (t == S_ - 1)
                out[(size_t)B_ * S_ * H_ + (size_t)b * H_ + ng] = v;
        }

        __syncthreads();   // step boundary: nxt complete, cur free for reuse
    }
}

// ---------------------------------------------------------------- launcher
extern "C" void kernel_launch(void* const* d_in, const int* in_sizes, int n_in,
                              void* d_out, int out_size, void* d_ws, size_t ws_size,
                              hipStream_t stream) {
    const float* x   = (const float*)d_in[0];   // [B,S,I]
    const float* h0  = (const float*)d_in[1];   // [1,B,H]
    const float* wih = (const float*)d_in[2];   // [H,I]
    const float* whh = (const float*)d_in[3];   // [H,H]
    const float* bih = (const float*)d_in[4];   // [H]
    const float* bhh = (const float*)d_in[5];   // [H]
    float* out = (float*)d_out;                 // [B,S,H] ++ [1,B,H]

    char* ws = (char*)d_ws;
    bf16*  wih_bf = (bf16*)(ws + OFF_WIH);
    bf16*  whh_bf = (bf16*)(ws + OFF_WHH);
    bf16*  hb0    = (bf16*)(ws + OFF_HB0);
    float* bsum   = (float*)(ws + OFF_BSUM);

    rnn_prep<<<(H_ * I_ + 255) / 256, 256, 0, stream>>>(
        wih, whh, h0, bih, bhh, wih_bf, whh_bf, hb0, bsum);

    rnn_xproj<<<dim3(H_ / 128, (B_ * S_) / 128), 256, 0, stream>>>(
        x, wih_bf, bsum, out);

    rnn_scan<<<B_ / 16, 1024, 0, stream>>>(out, whh_bf, hb0);
}
